// MyLayer_53489522704524
// MI455X (gfx1250) — matrix operand order repaired and checked by reading, not compiled, not run on hardware
//
#include <hip/hip_runtime.h>

typedef __attribute__((ext_vector_type(16))) _Float16 v16h;
typedef __attribute__((ext_vector_type(8)))  float    v8f;

#define VOCAB  5000
#define EMB    16
#define HID    32
#define NG     128      // 4*HID gate columns
#define BATCH  256
#define SEQ    2048
#define TILE_M 16       // batch rows per block = one WMMA M tile
#define ZPITCH 132      // padded row pitch (floats) for the z tile in LDS

// Fast transcendental path: v_exp_f32 + v_rcp_f32 (both TRANS ops that
// co-execute with VALU on CDNA5) instead of precise IEEE division
// (v_div_scale + Newton chain) that the compiler emits for '/'.
__device__ __forceinline__ float fast_sigmoid(float x) {
  return __builtin_amdgcn_rcpf(1.0f + __expf(-x));
}
__device__ __forceinline__ float fast_tanh(float x) {
  return 1.0f - 2.0f * __builtin_amdgcn_rcpf(__expf(2.0f * x) + 1.0f);
}

// Fused embedding + input-projection + LSTM recurrence.
// grid = 16 blocks (batch tiles of 16 rows), block = 256 threads = 8 waves.
// Wave w computes gate columns [16w, 16w+16) via two v_wmma_f32_16x16x32_f16
// per timestep:  z = [x | pad] @ [Wk; 0]  +  h @ Wr  + b.
__global__ __launch_bounds__(256, 1) void lstm_fused_wmma(
    const int*   __restrict__ tokens,  // [B, T]
    const float* __restrict__ emb,     // [VOCAB, EMB]
    const float* __restrict__ Wk,      // [EMB, NG]
    const float* __restrict__ Wr,      // [HID, NG]
    const float* __restrict__ bias,    // [NG]
    float*       __restrict__ out)     // [B, T, HID]
{
  __shared__ float z_lds[TILE_M * ZPITCH];            // 8.4 KB gate pre-acts
  __shared__ __align__(32) _Float16 h_frag[32][16];   // h in WMMA-A layout
  __shared__ int tok_lds[2][TILE_M];                  // double-buffered tokens

  const int tid  = threadIdx.x;
  const int wave = tid >> 5;
  const int lane = tid & 31;
  const int l15  = lane & 15;
  const int hi   = lane >> 4;          // lane half selects K range
  const int col  = (wave << 4) + l15;  // gate column 0..127 (also N index)
  const int row0 = blockIdx.x * TILE_M;

  // ---- resident B fragments (f16), built once ----
  // 16-bit B 32x16: lanes 0-15 hold K=0..7,16..23; lanes 16-31 hold K=8..15,24..31.
  v16h b_wk = {};   // Wk padded: rows 16..31 are zero
  v16h b_wr;
#pragma unroll
  for (int k = 0; k < 16; ++k) {
    const int kk = (hi << 3) + k + ((k >> 3) << 3);
    if (kk < EMB) b_wk[k] = (_Float16)Wk[kk * NG + col];
    b_wr[k] = (_Float16)Wr[kk * NG + col];
  }
  const float bcol = bias[col];

  // ---- init state ----
  if (tid < 32) {
#pragma unroll
    for (int k = 0; k < 16; ++k) h_frag[tid][k] = (_Float16)0.0f;
  }
  if (tid < TILE_M) tok_lds[0][tid] = tokens[(size_t)(row0 + tid) * SEQ];

  // update-phase ownership: thread handles (m0, jj) and (m0+8, jj)
  const int jj = tid & 31;
  const int m0 = tid >> 5;
  float c_st0 = 0.0f, c_st1 = 0.0f;   // cell state lives in registers
  float h_st0 = 0.0f, h_st1 = 0.0f;

  __syncthreads();

  for (int t = 0; t < SEQ; ++t) {
    const int buf = t & 1;

    // ---- A fragments (16-bit A 16x32: lane row = lane&15;
    //      lanes 0-15 hold K=0..7,16..23; lanes 16-31 hold K=8..15,24..31) ----
    const int tokm = tok_lds[buf][l15];
    const float* erow = emb + (size_t)tokm * EMB + (hi << 3);
    v16h a_x = {};                       // K>=16 stays zero (padding)
#pragma unroll
    for (int k = 0; k < 8; ++k) a_x[k] = (_Float16)erow[k];
    v16h a_h = *(const v16h*)&h_frag[lane][0];   // pre-laid-out fragment

    v8f acc;
#pragma unroll
    for (int r = 0; r < 8; ++r) acc[r] = bcol;
    acc = __builtin_amdgcn_wmma_f32_16x16x32_f16(false, a_x, false, b_wk,
                                                 (short)0, acc, false, false);
    acc = __builtin_amdgcn_wmma_f32_16x16x32_f16(false, a_h, false, b_wr,
                                                 (short)0, acc, false, false);

    // D layout: lanes 0-15 -> M=r, lanes 16-31 -> M=r+8, N=col
#pragma unroll
    for (int r = 0; r < 8; ++r)
      z_lds[(r + (hi << 3)) * ZPITCH + col] = acc[r];

    __syncthreads();

    // ---- gates + state update ----
    {
      const float* zr0 = &z_lds[m0 * ZPITCH];
      float i0 = fast_sigmoid(zr0[jj]);
      float f0 = fast_sigmoid(zr0[HID + jj]);
      float g0 = fast_tanh   (zr0[2 * HID + jj]);
      float o0 = fast_sigmoid(zr0[3 * HID + jj]);
      float cn0 = f0 * c_st0 + i0 * g0;
      float hn0 = o0 * fast_tanh(cn0);
      if (tok_lds[buf][m0] != 0) { c_st0 = cn0; h_st0 = hn0; }

      const float* zr1 = &z_lds[(m0 + 8) * ZPITCH];
      float i1 = fast_sigmoid(zr1[jj]);
      float f1 = fast_sigmoid(zr1[HID + jj]);
      float g1 = fast_tanh   (zr1[2 * HID + jj]);
      float o1 = fast_sigmoid(zr1[3 * HID + jj]);
      float cn1 = f1 * c_st1 + i1 * g1;
      float hn1 = o1 * fast_tanh(cn1);
      if (tok_lds[buf][m0 + 8] != 0) { c_st1 = cn1; h_st1 = hn1; }

      // scatter h back into the WMMA A-fragment layout:
      //   element (m, j):  lane = m + 16*((j>>3)&1), half = (j&7) + 8*(j>>4)
      const int fl = (jj >> 3) & 1;
      const int fi = (jj & 7) + ((jj >> 4) << 3);
      h_frag[m0 + (fl << 4)][fi]     = (_Float16)h_st0;
      h_frag[m0 + 8 + (fl << 4)][fi] = (_Float16)h_st1;

      out[((size_t)(row0 + m0)     * SEQ + t) * HID + jj] = h_st0;
      out[((size_t)(row0 + m0 + 8) * SEQ + t) * HID + jj] = h_st1;
    }

    // prefetch next step's tokens into the other buffer
    if (tid < TILE_M && (t + 1) < SEQ)
      tok_lds[buf ^ 1][tid] = tokens[(size_t)(row0 + tid) * SEQ + t + 1];

    __syncthreads();
  }
}

extern "C" void kernel_launch(void* const* d_in, const int* in_sizes, int n_in,
                              void* d_out, int out_size, void* d_ws, size_t ws_size,
                              hipStream_t stream) {
  (void)in_sizes; (void)n_in; (void)out_size; (void)d_ws; (void)ws_size;
  const int*   tokens = (const int*)  d_in[0];
  const float* emb    = (const float*)d_in[1];
  const float* Wk     = (const float*)d_in[2];
  const float* Wr     = (const float*)d_in[3];
  const float* b      = (const float*)d_in[4];
  float* out = (float*)d_out;

  dim3 grid(BATCH / TILE_M);   // 16 batch tiles
  dim3 block(256);             // 8 waves: one 16-col gate slice each
  hipLaunchKernelGGL(lstm_fused_wmma, grid, block, 0, stream,
                     tokens, emb, Wk, Wr, b, out);
}
